// SwinTransformer_21586505630112
// MI455X (gfx1250) — matrix operand order repaired and checked by reading, not compiled, not run on hardware
//
#include <hip/hip_runtime.h>

// ---------------------------------------------------------------------------
// CDNA5 (gfx1250) Swin block, 3 kernels:
//   prep        : weights fp32->f16 (+q pre-scale), rel-pos bias table
//   swin_attn   : per-window fused LN1+shift+partition | QKV WMMA-GEMM (->LDS)
//                 | 6-head window attention (all operands in LDS) | proj+res
//   mlp         : fused LN2 + MLP1 + GELU + MLP2 + residual (h tile in LDS)
// All matmuls via v_wmma_f32_16x16x32_f16; all intermediates live in the
// 320KB/WGP LDS (144KB + 60KB tiles), HBM traffic ~0.6 GB (~26us @ 23.3TB/s).
// ---------------------------------------------------------------------------

typedef __attribute__((ext_vector_type(16))) _Float16 v16h;
typedef __attribute__((ext_vector_type(8)))  _Float16 v8h;
typedef __attribute__((ext_vector_type(8)))  float    v8f;

#define WMMA_F16(a, b, c) \
  __builtin_amdgcn_wmma_f32_16x16x32_f16(false, (a), false, (b), (short)0, (c), false, false)

static __device__ __forceinline__ v16h ld16(const _Float16* p0, const _Float16* p1) {
  v8h lo = *(const v8h*)p0;
  v8h hi = *(const v8h*)p1;
  return __builtin_shufflevector(lo, hi, 0,1,2,3,4,5,6,7,8,9,10,11,12,13,14,15);
}
static __device__ __forceinline__ unsigned pack2(float a, float b) {
  unsigned short l = __builtin_bit_cast(unsigned short, (_Float16)a);
  unsigned short h = __builtin_bit_cast(unsigned short, (_Float16)b);
  return (unsigned)l | ((unsigned)h << 16);
}

// ---------------------------------------------------------------------------
// Kernel 0: weights fp32 -> f16 (+ q pre-scale) and rel-pos bias table
// layout in w16: qkv[0..110591] proj[110592..] w1[147456..] w2[294912..]
// ---------------------------------------------------------------------------
__global__ __launch_bounds__(256) void prep_kernel(
    const float* __restrict__ qkvw, const float* __restrict__ projw,
    const float* __restrict__ w1,   const float* __restrict__ w2,
    const float* __restrict__ rpb,
    _Float16* __restrict__ w16, float* __restrict__ biasTab) {
  int i = blockIdx.x * 256 + threadIdx.x;
  if (i < 110592) {                         // qkv_w [576,192]
    int n = i / 192;
    float s = (n < 192) ? 0.17677669529663687f : 1.0f;  // hd^-0.5 on q rows
    w16[i] = (_Float16)(qkvw[i] * s);
    return;
  }
  i -= 110592;
  if (i < 36864) { w16[110592 + i] = (_Float16)projw[i]; return; }
  i -= 36864;
  if (i < 147456) { w16[147456 + i] = (_Float16)w1[i]; return; }
  i -= 147456;
  if (i < 147456) { w16[294912 + i] = (_Float16)w2[i]; return; }
  i -= 147456;
  if (i < 24576) {                          // biasTab[h][i64][j64]
    int h = i >> 12; int ij = i & 4095;
    int ii = ij >> 6, jj = ij & 63;
    int ri = ii >> 3, ci = ii & 7, rj = jj >> 3, cj = jj & 7;
    int idx = (ri - rj + 7) * 15 + (ci - cj + 7);
    biasTab[i] = rpb[idx * 6 + h];
  }
}

// ---------------------------------------------------------------------------
// Fused per-window kernel: 1 block = 1 window (64 tokens), 192 thr = 6 waves.
// LDS map (halves):
//   [0)      q   : head h at h*2048, q[t*32+d]        (12288)
//   [12288)  k   : head h at h*2048, k[t*32+d]        (12288)
//   [24576)  vT  : head h at h*2048, vT[d*64+t]       (12288)
//   [36864)  P   : wave w at w*4096, P[i*64+j]        (24576)
//   [61440)  xn / attn-out tile, [t*192+c]            (12288)   total 144 KiB
// ---------------------------------------------------------------------------
__global__ __launch_bounds__(192) void swin_attn_kernel(
    const float* __restrict__ x, const float* __restrict__ g1,
    const float* __restrict__ b1,
    const _Float16* __restrict__ Wqkv, const _Float16* __restrict__ Wproj,
    const float* __restrict__ proj_b, const float* __restrict__ biasTab,
    float* __restrict__ x1) {
  __shared__ _Float16 smem[73728];          // 147456 B
  _Float16* lq  = smem;
  _Float16* lk  = smem + 12288;
  _Float16* lvt = smem + 24576;
  _Float16* lP  = smem + 36864;
  _Float16* lxn = smem + 61440;             // aliased: LN tile -> attn-out tile

  int lane = threadIdx.x & 31, wid = threadIdx.x >> 5;
  int l16 = lane & 15, hf = lane >> 4;
  int w = blockIdx.x;
  int b_ = w >> 10, wi = w & 1023, wh = wi >> 5, ww = wi & 31;
  bool lastH = (wh == 31), lastW = (ww == 31);
  v8f zf = {};

  // ---- stage 0: LN1 + roll(-4,-4) gather, waves strided over tokens -----
  for (int t = wid; t < 64; t += 6) {
    int r8 = t >> 3, c8 = t & 7;
    int h2 = (wh * 8 + r8 + 4) & 255;
    int w2 = (ww * 8 + c8 + 4) & 255;
    size_t src = (((size_t)(b_ * 256 + h2)) * 256 + w2) * 192;
    float v[6];
    float s = 0.f;
#pragma unroll
    for (int k = 0; k < 6; ++k) { v[k] = x[src + lane + k * 32]; s += v[k]; }
    for (int o = 16; o; o >>= 1) s += __shfl_xor(s, o, 32);
    float mean = s * (1.0f / 192.0f);
    float vs = 0.f;
#pragma unroll
    for (int k = 0; k < 6; ++k) { float d = v[k] - mean; vs += d * d; }
    for (int o = 16; o; o >>= 1) vs += __shfl_xor(vs, o, 32);
    float rstd = rsqrtf(vs * (1.0f / 192.0f) + 1e-5f);
#pragma unroll
    for (int k = 0; k < 6; ++k) {
      int c = lane + k * 32;
      lxn[t * 192 + c] = (_Float16)((v[k] - mean) * rstd * g1[c] + b1[c]);
    }
  }
  __syncthreads();

  // ---- stage 1: QKV GEMM 64x576 (K=192), 18 wave-tiles of 32x64 ---------
#pragma unroll
  for (int q = 0; q < 3; ++q) {
    int id = wid + 6 * q;                   // 0..17
    int m0 = (id & 1) * 32;
    int n0 = (id >> 1) * 64;
    v8f acc[2][4];
#pragma unroll
    for (int mi = 0; mi < 2; ++mi)
#pragma unroll
      for (int ni = 0; ni < 4; ++ni) acc[mi][ni] = zf;
#pragma unroll
    for (int k0 = 0; k0 < 192; k0 += 32) {
      v16h a[2], b[4];
#pragma unroll
      for (int mi = 0; mi < 2; ++mi) {
        const _Float16* p = lxn + (m0 + mi * 16 + l16) * 192 + k0 + hf * 16;
        a[mi] = ld16(p, p + 8);
      }
#pragma unroll
      for (int ni = 0; ni < 4; ++ni) {
        const _Float16* p = Wqkv + (size_t)(n0 + ni * 16 + l16) * 192 + k0 + hf * 16;
        b[ni] = ld16(p, p + 8);
      }
#pragma unroll
      for (int mi = 0; mi < 2; ++mi)
#pragma unroll
        for (int ni = 0; ni < 4; ++ni)
          acc[mi][ni] = WMMA_F16(a[mi], b[ni], acc[mi][ni]);
    }
#pragma unroll
    for (int mi = 0; mi < 2; ++mi)
#pragma unroll
      for (int ni = 0; ni < 4; ++ni) {
        int col = n0 + ni * 16 + l16;
        int which = col / 192, cc = col % 192;
        int hh = cc >> 5, d = cc & 31;
#pragma unroll
        for (int r = 0; r < 8; ++r) {
          int t = m0 + mi * 16 + r + hf * 8;
          _Float16 hv = (_Float16)acc[mi][ni][r];
          if (which == 0)      lq[hh * 2048 + t * 32 + d] = hv;   // q (pre-scaled)
          else if (which == 1) lk[hh * 2048 + t * 32 + d] = hv;   // k  [64,32]
          else                 lvt[hh * 2048 + d * 64 + t] = hv;  // vT [32,64]
        }
      }
  }
  __syncthreads();

  // ---- stage 2: attention, wave == head ---------------------------------
  {
    int hh = wid;
    const _Float16* qb = lq + hh * 2048;
    const _Float16* kb = lk + hh * 2048;
    const _Float16* vtb = lvt + hh * 2048;
    const float* bias = biasTab + hh * 4096;

    // S^T[j,i] = sum_d k[j,d] q[i,d]
    v8f S[4][4];
    {
      v16h a[4], bq[4];
#pragma unroll
      for (int jt = 0; jt < 4; ++jt) {
        const _Float16* p = kb + (jt * 16 + l16) * 32 + hf * 16;
        a[jt] = ld16(p, p + 8);
      }
#pragma unroll
      for (int it = 0; it < 4; ++it) {
        const _Float16* p = qb + (it * 16 + l16) * 32 + hf * 16;
        bq[it] = ld16(p, p + 8);
      }
#pragma unroll
      for (int jt = 0; jt < 4; ++jt)
#pragma unroll
        for (int it = 0; it < 4; ++it)
          S[jt][it] = WMMA_F16(a[jt], bq[it], zf);
    }

    // bias + shift-window mask (on-the-fly region ids)
    int cnti[4];
#pragma unroll
    for (int it = 0; it < 4; ++it) {
      int i = it * 16 + l16;
      int r8 = i >> 3, c8 = i & 7;
      int hr = lastH ? (r8 < 4 ? 1 : 2) : 0;
      int wr = lastW ? (c8 < 4 ? 1 : 2) : 0;
      cnti[it] = hr * 3 + wr;
    }
#pragma unroll
    for (int jt = 0; jt < 4; ++jt) {
#pragma unroll
      for (int r = 0; r < 8; ++r) {
        int j = jt * 16 + r + hf * 8;
        int r8 = j >> 3, c8 = j & 7;
        int hr = lastH ? (r8 < 4 ? 1 : 2) : 0;
        int wr = lastW ? (c8 < 4 ? 1 : 2) : 0;
        int cj = hr * 3 + wr;
#pragma unroll
        for (int it = 0; it < 4; ++it) {
          int i = it * 16 + l16;
          float m = (cj != cnti[it]) ? -100.0f : 0.0f;
          S[jt][it][r] += bias[i * 64 + j] + m;
        }
      }
    }

    // softmax over j (VGPR rows + partner lane via shfl_xor 16)
#pragma unroll
    for (int it = 0; it < 4; ++it) {
      float mx = -1e30f;
#pragma unroll
      for (int jt = 0; jt < 4; ++jt)
#pragma unroll
        for (int r = 0; r < 8; ++r) mx = fmaxf(mx, S[jt][it][r]);
      mx = fmaxf(mx, __shfl_xor(mx, 16, 32));
      float sm = 0.f;
#pragma unroll
      for (int jt = 0; jt < 4; ++jt)
#pragma unroll
        for (int r = 0; r < 8; ++r) {
          float e = __expf(S[jt][it][r] - mx);
          S[jt][it][r] = e; sm += e;
        }
      sm += __shfl_xor(sm, 16, 32);
      float inv = 1.0f / sm;
#pragma unroll
      for (int jt = 0; jt < 4; ++jt)
#pragma unroll
        for (int r = 0; r < 8; ++r) S[jt][it][r] *= inv;
    }

    // stage P[i][j] in this wave's LDS slice
    _Float16* Pl = lP + wid * 4096;
    unsigned* Pl32 = (unsigned*)Pl;
#pragma unroll
    for (int jt = 0; jt < 4; ++jt)
#pragma unroll
      for (int it = 0; it < 4; ++it)
#pragma unroll
        for (int r = 0; r < 8; r += 2) {
          int i = it * 16 + l16;
          int jb = jt * 16 + hf * 8 + r;
          Pl32[(i * 64 + jb) >> 1] = pack2(S[jt][it][r], S[jt][it][r + 1]);
        }

    // O^T[d,i] = sum_j vT[d,j] * P[i,j]
    v8f O[2][4];
#pragma unroll
    for (int mt = 0; mt < 2; ++mt)
#pragma unroll
      for (int it = 0; it < 4; ++it) O[mt][it] = zf;
#pragma unroll
    for (int kc = 0; kc < 2; ++kc) {
      v16h av[2], bp[4];
#pragma unroll
      for (int mt = 0; mt < 2; ++mt) {
        const _Float16* p = vtb + (mt * 16 + l16) * 64 + kc * 32 + hf * 16;
        av[mt] = ld16(p, p + 8);
      }
#pragma unroll
      for (int it = 0; it < 4; ++it) {
        const _Float16* p = Pl + (it * 16 + l16) * 64 + kc * 32 + hf * 16;
        bp[it] = ld16(p, p + 8);
      }
#pragma unroll
      for (int mt = 0; mt < 2; ++mt)
#pragma unroll
        for (int it = 0; it < 4; ++it)
          O[mt][it] = WMMA_F16(av[mt], bp[it], O[mt][it]);
    }
    __syncthreads();   // all stage-1 readers done; lxn now reused as attn-out

    // attn-out tile [t][hh*32+d] into LDS (aliases lxn)
    unsigned* ao32 = (unsigned*)lxn;
#pragma unroll
    for (int mt = 0; mt < 2; ++mt)
#pragma unroll
      for (int it = 0; it < 4; ++it)
#pragma unroll
        for (int r = 0; r < 8; r += 2) {
          int i = it * 16 + l16;
          int db = mt * 16 + hf * 8 + r;
          ao32[(i * 192 + hh * 32 + db) >> 1] = pack2(O[mt][it][r], O[mt][it][r + 1]);
        }
  }
  __syncthreads();

  // ---- stage 3: proj GEMM 64x192 (K=192) + bias + reverse/roll + res ----
  {
    int m0 = (wid & 1) * 32;
    int n0 = (wid >> 1) * 64;                // wid 0..5 -> n0 0/64/128
    v8f acc[2][4];
#pragma unroll
    for (int mi = 0; mi < 2; ++mi)
#pragma unroll
      for (int ni = 0; ni < 4; ++ni) acc[mi][ni] = zf;
#pragma unroll
    for (int k0 = 0; k0 < 192; k0 += 32) {
      v16h a[2], b[4];
#pragma unroll
      for (int mi = 0; mi < 2; ++mi) {
        const _Float16* p = lxn + (m0 + mi * 16 + l16) * 192 + k0 + hf * 16;
        a[mi] = ld16(p, p + 8);
      }
#pragma unroll
      for (int ni = 0; ni < 4; ++ni) {
        const _Float16* p = Wproj + (size_t)(n0 + ni * 16 + l16) * 192 + k0 + hf * 16;
        b[ni] = ld16(p, p + 8);
      }
#pragma unroll
      for (int mi = 0; mi < 2; ++mi)
#pragma unroll
        for (int ni = 0; ni < 4; ++ni)
          acc[mi][ni] = WMMA_F16(a[mi], b[ni], acc[mi][ni]);
    }
#pragma unroll
    for (int mi = 0; mi < 2; ++mi)
#pragma unroll
      for (int ni = 0; ni < 4; ++ni) {
        int col = n0 + ni * 16 + l16;
#pragma unroll
        for (int r = 0; r < 8; ++r) {
          int t = m0 + mi * 16 + r + hf * 8;
          int r8 = t >> 3, c8 = t & 7;
          int h2 = (wh * 8 + r8 + 4) & 255;  // roll(+4) back
          int w2 = (ww * 8 + c8 + 4) & 255;
          size_t addr = (((size_t)(b_ * 256 + h2)) * 256 + w2) * 192 + col;
          x1[addr] = x[addr] + acc[mi][ni][r] + proj_b[col];
        }
      }
  }
}

// ---------------------------------------------------------------------------
// Fused LN2 + MLP1 + GELU + MLP2 + residual. One block = 32 tokens.
// ---------------------------------------------------------------------------
__global__ __launch_bounds__(128) void mlp_kernel(
    const float* __restrict__ x1, const float* __restrict__ g,
    const float* __restrict__ bln,
    const _Float16* __restrict__ W1, const _Float16* __restrict__ W2,
    const float* __restrict__ b1, const float* __restrict__ b2,
    float* __restrict__ out) {
  __shared__ _Float16 xn[32 * 192];     // 12 KiB
  __shared__ _Float16 hbuf[32 * 768];   // 48 KiB
  int lane = threadIdx.x & 31, wid = threadIdx.x >> 5;
  int l16 = lane & 15, hf = lane >> 4;
  int m0 = blockIdx.x * 32;

  // LN2: each wave normalizes 8 rows into LDS
#pragma unroll
  for (int r = 0; r < 8; ++r) {
    int lrow = wid * 8 + r;
    size_t src = (size_t)(m0 + lrow) * 192;
    float v[6];
    float s = 0.f;
#pragma unroll
    for (int k = 0; k < 6; ++k) { v[k] = x1[src + lane + k * 32]; s += v[k]; }
    for (int o = 16; o; o >>= 1) s += __shfl_xor(s, o, 32);
    float mean = s * (1.0f / 192.0f);
    float vs = 0.f;
#pragma unroll
    for (int k = 0; k < 6; ++k) { float d = v[k] - mean; vs += d * d; }
    for (int o = 16; o; o >>= 1) vs += __shfl_xor(vs, o, 32);
    float rstd = rsqrtf(vs * (1.0f / 192.0f) + 1e-5f);
#pragma unroll
    for (int k = 0; k < 6; ++k) {
      int c = lane + k * 32;
      xn[lrow * 192 + c] = (_Float16)((v[k] - mean) * rstd * g[c] + bln[c]);
    }
  }
  __syncthreads();

  // MLP1 + GELU: 12 n-tiles of 64, 3 per wave; A from LDS
  v8f zf = {};
#pragma unroll
  for (int q = 0; q < 3; ++q) {
    int n0 = (wid + 4 * q) * 64;
    v8f acc[2][4];
#pragma unroll
    for (int mi = 0; mi < 2; ++mi)
#pragma unroll
      for (int ni = 0; ni < 4; ++ni) acc[mi][ni] = zf;
#pragma unroll
    for (int k0 = 0; k0 < 192; k0 += 32) {
      v16h a[2], b[4];
#pragma unroll
      for (int mi = 0; mi < 2; ++mi) {
        const _Float16* p = xn + (mi * 16 + l16) * 192 + k0 + hf * 16;
        a[mi] = ld16(p, p + 8);
      }
#pragma unroll
      for (int ni = 0; ni < 4; ++ni) {
        const _Float16* p = W1 + (size_t)(n0 + ni * 16 + l16) * 192 + k0 + hf * 16;
        b[ni] = ld16(p, p + 8);
      }
#pragma unroll
      for (int mi = 0; mi < 2; ++mi)
#pragma unroll
        for (int ni = 0; ni < 4; ++ni)
          acc[mi][ni] = WMMA_F16(a[mi], b[ni], acc[mi][ni]);
    }
#pragma unroll
    for (int mi = 0; mi < 2; ++mi)
#pragma unroll
      for (int ni = 0; ni < 4; ++ni) {
        int col = n0 + ni * 16 + l16;
#pragma unroll
        for (int r = 0; r < 8; ++r) {
          int lrow = mi * 16 + r + hf * 8;
          float v = acc[mi][ni][r] + b1[col];
          v = 0.5f * v * (1.0f + erff(v * 0.70710678118654752f));  // exact GELU
          hbuf[lrow * 768 + col] = (_Float16)v;
        }
      }
  }
  __syncthreads();

  // MLP2: 3 n-tiles of 64 (waves 0..2), K=768 from LDS
  if (wid < 3) {
    int n0 = wid * 64;
    v8f acc[2][4];
#pragma unroll
    for (int mi = 0; mi < 2; ++mi)
#pragma unroll
      for (int ni = 0; ni < 4; ++ni) acc[mi][ni] = zf;
#pragma unroll 4
    for (int k0 = 0; k0 < 768; k0 += 32) {
      v16h a[2], b[4];
#pragma unroll
      for (int mi = 0; mi < 2; ++mi) {
        const _Float16* p = hbuf + (mi * 16 + l16) * 768 + k0 + hf * 16;
        a[mi] = ld16(p, p + 8);
      }
#pragma unroll
      for (int ni = 0; ni < 4; ++ni) {
        const _Float16* p = W2 + (size_t)(n0 + ni * 16 + l16) * 768 + k0 + hf * 16;
        b[ni] = ld16(p, p + 8);
      }
#pragma unroll
      for (int mi = 0; mi < 2; ++mi)
#pragma unroll
        for (int ni = 0; ni < 4; ++ni)
          acc[mi][ni] = WMMA_F16(a[mi], b[ni], acc[mi][ni]);
    }
#pragma unroll
    for (int mi = 0; mi < 2; ++mi)
#pragma unroll
      for (int ni = 0; ni < 4; ++ni) {
        int col = n0 + ni * 16 + l16;
#pragma unroll
        for (int r = 0; r < 8; ++r) {
          int row = m0 + mi * 16 + r + hf * 8;
          size_t addr = (size_t)row * 192 + col;
          out[addr] = x1[addr] + acc[mi][ni][r] + b2[col];
        }
      }
  }
}

// ---------------------------------------------------------------------------
// Host-side orchestration. Workspace (bytes):
//   [0)       w16 weights (884736)
//   [884736)  biasTab     (98304)
//   [1 MiB)   x1 fp32     (100663296)          total ~102 MB
// ---------------------------------------------------------------------------
extern "C" void kernel_launch(void* const* d_in, const int* in_sizes, int n_in,
                              void* d_out, int out_size, void* d_ws, size_t ws_size,
                              hipStream_t stream) {
  (void)in_sizes; (void)n_in; (void)out_size; (void)ws_size;
  const float* x       = (const float*)d_in[0];
  const float* n1g     = (const float*)d_in[1];
  const float* n1b     = (const float*)d_in[2];
  const float* qkv_w   = (const float*)d_in[3];
  const float* rpb     = (const float*)d_in[4];
  const float* proj_w  = (const float*)d_in[5];
  const float* proj_b  = (const float*)d_in[6];
  const float* n2g     = (const float*)d_in[7];
  const float* n2b     = (const float*)d_in[8];
  const float* mlp_w1  = (const float*)d_in[9];
  const float* mlp_b1  = (const float*)d_in[10];
  const float* mlp_w2  = (const float*)d_in[11];
  const float* mlp_b2  = (const float*)d_in[12];

  char* ws = (char*)d_ws;
  _Float16* w16     = (_Float16*)ws;
  float*    biasTab = (float*)(ws + 884736);
  float*    x1      = (float*)(ws + (1ull << 20));

  _Float16* qkvw_h  = w16;
  _Float16* projw_h = w16 + 110592;
  _Float16* w1_h    = w16 + 147456;
  _Float16* w2_h    = w16 + 294912;

  // 0) weight conversion + bias table
  prep_kernel<<<1824, 256, 0, stream>>>(qkv_w, proj_w, mlp_w1, mlp_w2, rpb, w16, biasTab);
  // 1) fused LN1 + QKV + window attention + proj + residual -> x1
  swin_attn_kernel<<<2048, 192, 0, stream>>>(x, n1g, n1b, qkvw_h, projw_h,
                                             proj_b, biasTab, x1);
  // 2) fused LN2 + MLP1 + GELU + MLP2 + residual -> d_out
  mlp_kernel<<<4096, 128, 0, stream>>>(x1, n2g, n2b, w1_h, w2_h,
                                       mlp_b1, mlp_b2, (float*)d_out);
}
